// ContrastiveLoss_76776835384072
// MI455X (gfx1250) — matrix operand order repaired and checked by reading, not compiled, not run on hardware
//
#include <hip/hip_runtime.h>
#include <hip/hip_bf16.h>
#include <stdint.h>

// ---- problem constants (match reference setup_inputs) ----
#define B_    16
#define C_    256
#define H_    32
#define Wd_   32
#define KMAX  5
#define NEGS  16
#define SKIP_ 1
// rows in transposed layout: H_*Wd_*B_ = 16384
// maxN = (H_-SKIP_-1)*Wd_*B_ = 15360

typedef __attribute__((ext_vector_type(16))) __bf16 v16bf;
typedef __attribute__((ext_vector_type(8)))  float  v8f;

struct alignas(16) U4 { unsigned int a, b, c, d; };
union FragU { v16bf v; U4 q[2]; };

__device__ __forceinline__ unsigned short f2bf_rne(float x) {
  union { float f; unsigned int u; } v; v.f = x;
  unsigned int u = v.u;
  unsigned int r = (u + 0x7FFFu + ((u >> 16) & 1u)) >> 16;
  return (unsigned short)r;
}

// ---- build zrows (bf16) and crows (f32): row = (h*32+w)*16 + b, col = c ----
__global__ void transpose_zc(const float* __restrict__ z, const float* __restrict__ cin,
                             unsigned short* __restrict__ zb, float* __restrict__ cr) {
  int idx = blockIdx.x * blockDim.x + threadIdx.x;
  if (idx >= B_ * C_ * H_ * Wd_) return;
  int w = idx & 31;
  int h = (idx >> 5) & 31;
  int c = (idx >> 10) & 255;
  int b = idx >> 18;
  int row = (h * 32 + w) * 16 + b;
  zb[(size_t)row * 256 + c] = f2bf_rne(z[idx]);
  cr[(size_t)row * 256 + c] = cin[idx];
}

// ---- elementwise f32 -> bf16 copy of W (B fragment wants W[o][c], contiguous in c) ----
__global__ void convert_w(const float* __restrict__ Wsrc, unsigned short* __restrict__ wb, int n) {
  int i = blockIdx.x * blockDim.x + threadIdx.x;
  if (i < n) wb[i] = f2bf_rne(Wsrc[i]);
}

// ---- GEMM: flat[r][o] = sum_c A[r][c] * W[o][c], one wave per 16x16 tile ----
__global__ void gemm_wmma(const unsigned short* __restrict__ zb,
                          const unsigned short* __restrict__ wb,   // Wk bf16, [o][c]
                          float* __restrict__ flat,
                          int rowBase) {                           // (k+SKIP)*512
  int lane = threadIdx.x & 31;
  int half = lane >> 4;
  int m    = lane & 15;
  int rowTile = blockIdx.x;
  int o0      = blockIdx.y << 4;

  const unsigned short* Arow = zb + (size_t)(rowBase + rowTile * 16 + m) * 256;
  const unsigned short* Brow = wb + (size_t)(o0 + m) * 256;

  v8f acc = {};
  #pragma unroll
  for (int c0 = 0; c0 < 256; c0 += 32) {
    FragU a, bf;
    // A 16x32 bf16 layout: element e -> K = (e<8 ? 0 : 16) + 8*half + (e&7)
    a.q[0] = *(const U4*)(Arow + c0 + 8 * half);
    a.q[1] = *(const U4*)(Arow + c0 + 16 + 8 * half);
    // B 32x16 bf16 layout: element e -> K = (e<8 ? 0 : 16) + 8*half + (e&7), N = lane%16
    bf.q[0] = *(const U4*)(Brow + c0 + 8 * half);
    bf.q[1] = *(const U4*)(Brow + c0 + 16 + 8 * half);
    acc = __builtin_amdgcn_wmma_f32_16x16x32_bf16(false, a.v, false, bf.v,
                                                  (short)0, acc, false, false);
  }
  // D layout: VGPR i -> row M = i + 8*half, col N = lane%16
  float* out = flat + (size_t)rowTile * 16 * 256 + o0;
  #pragma unroll
  for (int i = 0; i < 8; ++i)
    out[(size_t)(i + 8 * half) * 256 + m] = acc[i];
}

// ---- scoring: one wave per position, 1 positive + 16 negative dots over C=256 ----
__global__ void score_kernel(const float* __restrict__ cr,
                             const float* __restrict__ flat,
                             const int*   __restrict__ idx,      // base for this k
                             float* __restrict__ partials,       // [nblocks][2]
                             int Nk) {
  __shared__ float sh[16];
  int lane = threadIdx.x & 31;
  int wid  = threadIdx.x >> 5;                 // 0..7
  int p    = blockIdx.x * 8 + wid;             // grid sized so p < Nk always

  const float* ctx = cr + (size_t)p * 256;
  float cv[8];
  #pragma unroll
  for (int j = 0; j < 8; ++j) cv[j] = ctx[lane + 32 * j];

  const float* zr = flat + (size_t)p * 256;
  float s = 0.f;
  #pragma unroll
  for (int j = 0; j < 8; ++j) s += cv[j] * zr[lane + 32 * j];
  #pragma unroll
  for (int off = 16; off > 0; off >>= 1) s += __shfl_xor(s, off, 32);
  float posLoss = (s < 1.0f) ? (1.0f - s) : 0.0f;

  float negLoss = 0.f;
  for (int n = 0; n < NEGS; ++n) {
    int jr = idx[p * NEGS + n] % Nk;
    const float* zn = flat + (size_t)jr * 256;
    float t = 0.f;
    #pragma unroll
    for (int j = 0; j < 8; ++j) t += cv[j] * zn[lane + 32 * j];
    #pragma unroll
    for (int off = 16; off > 0; off >>= 1) t += __shfl_xor(t, off, 32);
    if (t > -1.0f) negLoss += 1.0f + t;
  }

  if (lane == 0) { sh[wid] = posLoss; sh[8 + wid] = negLoss; }
  __syncthreads();
  if (threadIdx.x == 0) {
    float ps = 0.f, ns = 0.f;
    for (int i = 0; i < 8; ++i) { ps += sh[i]; ns += sh[8 + i]; }
    partials[(size_t)blockIdx.x * 2]     = ps;
    partials[(size_t)blockIdx.x * 2 + 1] = ns;
  }
}

// ---- deterministic final reduction over all per-block partials ----
__global__ void finalize_kernel(const float* __restrict__ partials, float* __restrict__ out) {
  __shared__ float red[256];
  float total = 0.f;
  int base = 0;
  for (int k = 1; k <= KMAX; ++k) {
    int y  = H_ - SKIP_ - k;
    int Nk = y * Wd_ * B_;
    int nb = Nk / 8;
    float ps = 0.f, ns = 0.f;
    for (int i = threadIdx.x; i < nb; i += blockDim.x) {
      ps += partials[(size_t)(base + i) * 2];
      ns += partials[(size_t)(base + i) * 2 + 1];
    }
    red[threadIdx.x] = ps; __syncthreads();
    for (int s = 128; s > 0; s >>= 1) {
      if (threadIdx.x < s) red[threadIdx.x] += red[threadIdx.x + s];
      __syncthreads();
    }
    float psum = red[0]; __syncthreads();
    red[threadIdx.x] = ns; __syncthreads();
    for (int s = 128; s > 0; s >>= 1) {
      if (threadIdx.x < s) red[threadIdx.x] += red[threadIdx.x + s];
      __syncthreads();
    }
    float nsum = red[0]; __syncthreads();
    if (threadIdx.x == 0)
      total += psum / (float)Nk + nsum / ((float)Nk * (float)NEGS);
    base += nb;
  }
  if (threadIdx.x == 0) out[0] = total;
}

extern "C" void kernel_launch(void* const* d_in, const int* in_sizes, int n_in,
                              void* d_out, int out_size, void* d_ws, size_t ws_size,
                              hipStream_t stream) {
  (void)in_sizes; (void)n_in; (void)out_size; (void)ws_size;
  const float* z    = (const float*)d_in[0];
  const float* cin  = (const float*)d_in[1];
  const float* Wm   = (const float*)d_in[2];
  const int*   ridx = (const int*)d_in[3];

  char* ws = (char*)d_ws;
  size_t off = 0;
  unsigned short* zb = (unsigned short*)(ws + off); off += (size_t)16384 * 256 * 2;   // 8 MB
  unsigned short* wb = (unsigned short*)(ws + off); off += (size_t)KMAX * 256 * 256 * 2;
  float* cr   = (float*)(ws + off); off += (size_t)16384 * 256 * 4;                   // 16 MB
  float* flat = (float*)(ws + off); off += (size_t)15360 * 256 * 4;                   // 15.7 MB
  float* partials = (float*)(ws + off);                                               // <= 72 KB

  int total = B_ * C_ * H_ * Wd_;
  transpose_zc<<<(total + 255) / 256, 256, 0, stream>>>(z, cin, zb, cr);
  int wn = KMAX * 256 * 256;
  convert_w<<<(wn + 255) / 256, 256, 0, stream>>>(Wm, wb, wn);

  const int ridxStride = 15360 * NEGS;  // maxN * NEG
  int base = 0;
  for (int k = 1; k <= KMAX; ++k) {
    int y  = H_ - SKIP_ - k;
    int Nk = y * Wd_ * B_;
    dim3 g(Nk / 16, 16);
    gemm_wmma<<<g, 32, 0, stream>>>(zb, wb + (size_t)(k - 1) * 256 * 256, flat,
                                    (k + SKIP_) * Wd_ * B_);
    int nb = Nk / 8;
    score_kernel<<<nb, 256, 0, stream>>>(cr, flat,
                                         ridx + (size_t)(k - 1) * ridxStride,
                                         partials + (size_t)base * 2, Nk);
    base += nb;
  }
  finalize_kernel<<<1, 256, 0, stream>>>(partials, (float*)d_out);
}
